// MambaSynapseEncoder_59923383714431
// MI455X (gfx1250) — compile-verified, matrix operand order checked by reading, not run on hardware
//
#include <hip/hip_runtime.h>
#include <hip/hip_bf16.h>
#include <stdint.h>

// Problem constants (fixed by the reference harness)
#define NTOK   16384
#define SEQLEN 1024
#define NSEQ   16
#define DIN    6
#define DMODEL 512
#define DINNER 1024
#define DSTATE 16
#define DCONV  4
#define DTRANK 32
#define XZN    (2*DINNER)   // 2048
#define DBCN   (DTRANK+2*DSTATE) // 64
#define NCHUNK 8
#define CLEN   (SEQLEN/NCHUNK) // 128

typedef __bf16 bf16_t;
typedef bf16_t v16bf __attribute__((ext_vector_type(16)));
typedef float  v8f   __attribute__((ext_vector_type(8)));

union Frag { v16bf v; uint4 q[2]; };

__device__ __forceinline__ unsigned short f2bf(float f) {
    unsigned int u = __float_as_uint(f);
    unsigned int r = u + 0x7FFFu + ((u >> 16) & 1u);
    return (unsigned short)(r >> 16);
}
__device__ __forceinline__ float silu(float x) { return x / (1.f + __expf(-x)); }

// ---------------- segment counts / starts / lengths (single block) ----------
__global__ void seg_kernel(const int* __restrict__ idx, int n,
                           int* __restrict__ counts, int* __restrict__ starts,
                           float* __restrict__ lengths) {
    __shared__ int cnt[NSEQ];
    int tid = threadIdx.x;
    if (tid < NSEQ) cnt[tid] = 0;
    __syncthreads();
    for (int i = tid; i < n; i += blockDim.x) atomicAdd(&cnt[idx[i]], 1);
    __syncthreads();
    if (tid == 0) {
        int run = 0;
        for (int b = 0; b < NSEQ; ++b) {
            int c = cnt[b];
            counts[b] = c; starts[b] = run; run += c;
            lengths[b] = (float)(c > 0 ? c : 1);
        }
    }
}

// ---------------- f32 -> bf16 convert (weights) ------------------------------
__global__ void cvt_bf16_kernel(const float* __restrict__ src,
                                unsigned short* __restrict__ dst, int n) {
    int i = blockIdx.x * blockDim.x + threadIdx.x;
    if (i < n) dst[i] = f2bf(src[i]);
}

// ---------------- x0 = dense @ W_in + b_in  (store bf16) ---------------------
__global__ void x0_kernel(const float* __restrict__ synapse,
                          const int* __restrict__ counts, const int* __restrict__ starts,
                          const float* __restrict__ W_in, const float* __restrict__ b_in,
                          unsigned short* __restrict__ x0bf) {
    int row = blockIdx.x;               // b*SEQLEN + t
    int b = row / SEQLEN, t = row - b * SEQLEN;
    bool valid = t < counts[b];
    float s[DIN];
    if (valid) {
        const float* sp = synapse + (size_t)(starts[b] + t) * DIN;
#pragma unroll
        for (int i = 0; i < DIN; ++i) s[i] = sp[i];
    }
    for (int c = threadIdx.x; c < DMODEL; c += blockDim.x) {
        float v = b_in[c];
        if (valid) {
#pragma unroll
            for (int i = 0; i < DIN; ++i) v = fmaf(s[i], W_in[i * DMODEL + c], v);
        }
        x0bf[(size_t)row * DMODEL + c] = f2bf(v);
    }
}

// ---------------- generic bf16 WMMA GEMM: C[M,N] = A[M,K] @ B[K,N] ----------
// Block tile 128(M) x 64(N), 8 waves, wave = 16x64 via 4 WMMA accumulators.
// B staging is software-pipelined: the next 16B chunk is issued right after
// the first barrier so HBM latency is hidden under the WMMA compute phase.
// epi: 0 = plain f32 store
//      1 = softplus(acc + bias[n]) f32 store
//      2 = plain f32 store + bf16 copy of cols < DTRANK into aux[m*DTRANK+n]
#define BS_STRIDE 40   // 40 bf16 = 80B row stride: 16B-aligned, bank-spread
__global__ void gemm_bf16_kernel(const unsigned short* __restrict__ Ag,
                                 const unsigned short* __restrict__ Bg,
                                 float* __restrict__ Cg,
                                 unsigned short* __restrict__ aux,
                                 const float* __restrict__ bias,
                                 int M, int N, int K, int epi) {
    __shared__ __align__(16) unsigned short Bs[64 * BS_STRIDE];
    const int tid  = threadIdx.x;
    const int wave = tid >> 5;
    const int lane = tid & 31;
    const int half = lane >> 4;
    const int nl   = lane & 15;
    const int m0 = blockIdx.x * 128 + wave * 16;
    const int n0 = blockIdx.y * 64;

    v8f acc[4];
#pragma unroll
    for (int j = 0; j < 4; ++j)
#pragma unroll
        for (int r = 0; r < 8; ++r) acc[j][r] = 0.f;

    // A-fragment source: two 16B chunks per lane, advanced by 32 bf16 per step
    const unsigned short* aptr = Ag + (size_t)(m0 + nl) * K + half * 8;

    // B staging: thread -> (row kl, 8 cols starting at nn) of the 32x64 tile
    const int kl = tid >> 3;           // 0..31
    const int nn = (tid & 7) * 8;      // 0,8,...,56
    const unsigned short* bsrc = Bg + (size_t)kl * N + n0 + nn;
    unsigned short* bdst = &Bs[nn * BS_STRIDE + kl];
    const size_t bstep = (size_t)32 * N;

    union { uint4 q; unsigned short h[8]; } bl;
    bl.q = *(const uint4*)bsrc;
    bsrc += bstep;

    for (int kk = 0; kk < K; kk += 32) {
#pragma unroll
        for (int j = 0; j < 8; ++j) bdst[j * BS_STRIDE] = bl.h[j];
        __syncthreads();

        if (kk + 32 < K) {                 // issue next tile load; waited at
            if (kk + 64 < K) __builtin_prefetch(bsrc + bstep, 0, 1);
            bl.q = *(const uint4*)bsrc;    // next iteration's ds_store
            bsrc += bstep;
        }

        Frag a;
        a.q[0] = *(const uint4*)(aptr);
        a.q[1] = *(const uint4*)(aptr + 16);
        aptr += 32;
#pragma unroll
        for (int j = 0; j < 4; ++j) {
            const unsigned short* brow = &Bs[(j * 16 + nl) * BS_STRIDE + half * 8];
            Frag bb;
            bb.q[0] = *(const uint4*)(brow);
            bb.q[1] = *(const uint4*)(brow + 16);
            acc[j] = __builtin_amdgcn_wmma_f32_16x16x32_bf16(
                false, a.v, false, bb.v, (short)0, acc[j], false, false);
        }
        __syncthreads();
    }

    // epilogue: lane element r -> C[m0 + half*8 + r][n0 + j*16 + nl]
#pragma unroll
    for (int j = 0; j < 4; ++j) {
        int n = n0 + j * 16 + nl;
#pragma unroll
        for (int r = 0; r < 8; ++r) {
            int m = m0 + half * 8 + r;
            float v = acc[j][r];
            if (epi == 1) {
                v += bias[n];
                v = (v > 20.f) ? v : __logf(1.f + __expf(v));
                Cg[(size_t)m * N + n] = v;
            } else {
                Cg[(size_t)m * N + n] = v;
                if (epi == 2 && n < DTRANK) aux[(size_t)m * DTRANK + n] = f2bf(v);
            }
        }
    }
}

// ---------------- depthwise conv (D_CONV=4) + SiLU ---------------------------
__global__ void conv_silu_kernel(const float* __restrict__ xz,
                                 const float* __restrict__ conv_w,
                                 const float* __restrict__ conv_b,
                                 float* __restrict__ xs_f32,
                                 unsigned short* __restrict__ xs_bf16) {
    int row = blockIdx.x;                   // b*SEQLEN + t
    int t = row & (SEQLEN - 1);
    int d = threadIdx.x * 4;
    float4 bv = *(const float4*)(conv_b + d);
    float acc[4] = {bv.x, bv.y, bv.z, bv.w};
#pragma unroll
    for (int k = 0; k < DCONV; ++k) {
        int tt = t - (DCONV - 1) + k;
        if (tt >= 0) {
            float4 xv = *(const float4*)(xz + (size_t)(row - (DCONV - 1) + k) * XZN + d);
            float4 wv = *(const float4*)(conv_w + k * DINNER + d);
            acc[0] = fmaf(xv.x, wv.x, acc[0]);
            acc[1] = fmaf(xv.y, wv.y, acc[1]);
            acc[2] = fmaf(xv.z, wv.z, acc[2]);
            acc[3] = fmaf(xv.w, wv.w, acc[3]);
        }
    }
    float4 out;
    out.x = silu(acc[0]); out.y = silu(acc[1]);
    out.z = silu(acc[2]); out.w = silu(acc[3]);
    *(float4*)(xs_f32 + (size_t)row * DINNER + d) = out;
    union { unsigned short s[4]; uint2 u; } pk;
    pk.s[0] = f2bf(out.x); pk.s[1] = f2bf(out.y);
    pk.s[2] = f2bf(out.z); pk.s[3] = f2bf(out.w);
    *(uint2*)(xs_bf16 + (size_t)row * DINNER + d) = pk.u;
}

// ---------------- chunk-parallel selective scan, pass 1 ----------------------
// The recurrence h_t = a_t*h_{t-1} + u_t is linear, so per chunk we compute:
//   h0[s]  : chunk-final state assuming zero entry state
//   E[s]   : prod of a_t over chunk (decay applied to entry state)
//   g[s]   : sum_t w_t*C_t[s]*P_t[s]  (pooled-sum sensitivity to entry state)
//   S0     : sum_t w_t*(C_t.h_t^0 + x_t*D)  (entry-state-independent part)
// where P_t = prod_{tau<=t} a_tau and w_t = silu(z_t).
__global__ void scan_chunk_kernel(const float* __restrict__ xz,
                                  const float* __restrict__ xs,
                                  const float* __restrict__ dbc,
                                  const float* __restrict__ delta,
                                  const float* __restrict__ A_log,
                                  const float* __restrict__ D_param,
                                  const int* __restrict__ counts,
                                  float* __restrict__ Earr,
                                  float* __restrict__ Harr,
                                  float* __restrict__ Garr,
                                  float* __restrict__ S0arr) {
    // grid.x = NSEQ * NCHUNK * (DINNER/256)
    int blk = blockIdx.x;
    int b = blk >> 5;                       // / (NCHUNK * 4)
    int c = (blk >> 2) & (NCHUNK - 1);
    int d = ((blk & 3) << 8) + threadIdx.x;

    float A[DSTATE], h[DSTATE], E[DSTATE], g[DSTATE];
#pragma unroll
    for (int s = 0; s < DSTATE; ++s) {
        A[s] = -__expf(A_log[d * DSTATE + s]);
        h[s] = 0.f; E[s] = 1.f; g[s] = 0.f;
    }
    const float Dp = D_param[d];
    const int len = counts[b];
    const int t0 = c * CLEN;
    const int t1 = (t0 + CLEN < len) ? (t0 + CLEN) : len;
    float S0 = 0.f;

    const size_t row0 = (size_t)b * SEQLEN + t0;
    const float* pdelta = delta + row0 * DINNER + d;
    const float* pxs    = xs    + row0 * DINNER + d;
    const float* pz     = xz    + row0 * XZN + DINNER + d;
    const float* pbc    = dbc   + row0 * DBCN + DTRANK;

    for (int t = t0; t < t1; ++t) {
        float dt = *pdelta;  pdelta += DINNER;
        float xv = *pxs;     pxs    += DINNER;
        float zv = *pz;      pz     += XZN;
        const float* BC = pbc;       pbc += DBCN;
        float w  = silu(zv);
        float dx = dt * xv;
        float y = 0.f;
#pragma unroll
        for (int s = 0; s < DSTATE; ++s) {
            float a = __expf(dt * A[s]);
            h[s] = a * h[s] + dx * BC[s];
            E[s] *= a;
            float Cs = BC[DSTATE + s];
            y = fmaf(h[s], Cs, y);
            g[s] = fmaf(w * Cs, E[s], g[s]);
        }
        S0 += w * fmaf(xv, Dp, y);
    }

    size_t base = ((size_t)(b * NCHUNK + c) * DINNER + d) * DSTATE;
#pragma unroll
    for (int s = 0; s < DSTATE; s += 4) {
        *(float4*)(Earr + base + s) = make_float4(E[s], E[s+1], E[s+2], E[s+3]);
        *(float4*)(Harr + base + s) = make_float4(h[s], h[s+1], h[s+2], h[s+3]);
        *(float4*)(Garr + base + s) = make_float4(g[s], g[s+1], g[s+2], g[s+3]);
    }
    S0arr[base / DSTATE] = S0;
}

// ---------------- chunk-parallel selective scan, pass 2 (combine) ------------
__global__ void scan_combine_kernel(const float* __restrict__ Earr,
                                    const float* __restrict__ Harr,
                                    const float* __restrict__ Garr,
                                    const float* __restrict__ S0arr,
                                    const float* __restrict__ lengths,
                                    float* __restrict__ ypool) {
    int i = blockIdx.x * blockDim.x + threadIdx.x;   // b*DINNER + d
    if (i >= NSEQ * DINNER) return;
    int b = i >> 10;
    int d = i & (DINNER - 1);
    float h[DSTATE];
#pragma unroll
    for (int s = 0; s < DSTATE; ++s) h[s] = 0.f;
    float total = 0.f;
    for (int c = 0; c < NCHUNK; ++c) {
        size_t base = ((size_t)(b * NCHUNK + c) * DINNER + d) * DSTATE;
        float dot = 0.f;
#pragma unroll
        for (int s = 0; s < DSTATE; ++s) dot = fmaf(Garr[base + s], h[s], dot);
        total += S0arr[base / DSTATE] + dot;
#pragma unroll
        for (int s = 0; s < DSTATE; ++s)
            h[s] = fmaf(Earr[base + s], h[s], Harr[base + s]);
    }
    ypool[i] = total / lengths[b];
}

// ---------------- small final GEMMs on pooled [16,*] -------------------------
__global__ void small_gemm_kernel(const float* __restrict__ A, // [16,K]
                                  const float* __restrict__ B, // [K,N]
                                  const float* __restrict__ bias,
                                  float* __restrict__ C, int N, int K) {
    int i = blockIdx.x * blockDim.x + threadIdx.x;
    if (i >= NSEQ * N) return;
    int r = i / N, c = i - r * N;
    float acc = bias ? bias[c] : 0.f;
    const float* a = A + (size_t)r * K;
    for (int k = 0; k < K; ++k) acc = fmaf(a[k], B[(size_t)k * N + c], acc);
    C[i] = acc;
}

// ---------------- host launcher ----------------------------------------------
static inline size_t align256(size_t x) { return (x + 255) & ~(size_t)255; }

extern "C" void kernel_launch(void* const* d_in, const int* in_sizes, int n_in,
                              void* d_out, int out_size, void* d_ws, size_t ws_size,
                              hipStream_t stream) {
    const float* synapse   = (const float*)d_in[0];
    const int*   syn_idx   = (const int*)d_in[1];
    const float* W_in      = (const float*)d_in[4];
    const float* b_in      = (const float*)d_in[5];
    const float* W_inproj  = (const float*)d_in[6];
    const float* conv_w    = (const float*)d_in[7];
    const float* conv_b    = (const float*)d_in[8];
    const float* W_xproj   = (const float*)d_in[9];
    const float* W_dt      = (const float*)d_in[10];
    const float* b_dt      = (const float*)d_in[11];
    const float* A_log     = (const float*)d_in[12];
    const float* D_param   = (const float*)d_in[13];
    const float* W_outproj = (const float*)d_in[14];
    const float* W_out     = (const float*)d_in[15];
    const float* b_out     = (const float*)d_in[16];
    float* out = (float*)d_out;

    // workspace carve-up
    char* ws = (char*)d_ws;
    size_t off = 0;
    int* counts = (int*)(ws + off);                 off = align256(off + NSEQ * 4);
    int* starts = (int*)(ws + off);                 off = align256(off + NSEQ * 4);
    float* lengths = (float*)(ws + off);            off = align256(off + NSEQ * 4);
    unsigned short* x0bf   = (unsigned short*)(ws + off); off = align256(off + (size_t)NTOK * DMODEL * 2);
    unsigned short* Wip_bf = (unsigned short*)(ws + off); off = align256(off + (size_t)DMODEL * XZN * 2);
    unsigned short* Wxp_bf = (unsigned short*)(ws + off); off = align256(off + (size_t)DINNER * DBCN * 2);
    unsigned short* Wdt_bf = (unsigned short*)(ws + off); off = align256(off + (size_t)DTRANK * DINNER * 2);
    float* xz      = (float*)(ws + off);            off = align256(off + (size_t)NTOK * XZN * 4);
    float* xs_f32  = (float*)(ws + off);            off = align256(off + (size_t)NTOK * DINNER * 4);
    unsigned short* xs_bf = (unsigned short*)(ws + off);  off = align256(off + (size_t)NTOK * DINNER * 2);
    float* dbc     = (float*)(ws + off);            off = align256(off + (size_t)NTOK * DBCN * 4);
    unsigned short* dtr_bf = (unsigned short*)(ws + off); off = align256(off + (size_t)NTOK * DTRANK * 2);
    float* delta   = (float*)(ws + off);            off = align256(off + (size_t)NTOK * DINNER * 4);
    size_t chunkN  = (size_t)NSEQ * NCHUNK * DINNER;
    float* Earr    = (float*)(ws + off);            off = align256(off + chunkN * DSTATE * 4);
    float* Harr    = (float*)(ws + off);            off = align256(off + chunkN * DSTATE * 4);
    float* Garr    = (float*)(ws + off);            off = align256(off + chunkN * DSTATE * 4);
    float* S0arr   = (float*)(ws + off);            off = align256(off + chunkN * 4);
    float* ypool   = (float*)(ws + off);            off = align256(off + (size_t)NSEQ * DINNER * 4);
    float* pooled  = (float*)(ws + off);            off = align256(off + (size_t)NSEQ * DMODEL * 4);
    (void)ws_size; (void)n_in; (void)out_size; (void)in_sizes;

    // 1) segment bookkeeping
    seg_kernel<<<1, 256, 0, stream>>>(syn_idx, NTOK, counts, starts, lengths);

    // 2) weight conversions to bf16
    cvt_bf16_kernel<<<(DMODEL * XZN + 255) / 256, 256, 0, stream>>>(W_inproj, Wip_bf, DMODEL * XZN);
    cvt_bf16_kernel<<<(DINNER * DBCN + 255) / 256, 256, 0, stream>>>(W_xproj, Wxp_bf, DINNER * DBCN);
    cvt_bf16_kernel<<<(DTRANK * DINNER + 255) / 256, 256, 0, stream>>>(W_dt, Wdt_bf, DTRANK * DINNER);

    // 3) x0 = dense @ W_in + b_in  (bf16)
    x0_kernel<<<NTOK, 256, 0, stream>>>(synapse, counts, starts, W_in, b_in, x0bf);

    // 4) xz = x0 @ W_inproj   (M=16384, N=2048, K=512)  -- WMMA
    gemm_bf16_kernel<<<dim3(NTOK / 128, XZN / 64), 256, 0, stream>>>(
        x0bf, Wip_bf, xz, nullptr, nullptr, NTOK, XZN, DMODEL, 0);

    // 5) depthwise conv + SiLU
    conv_silu_kernel<<<NTOK, 256, 0, stream>>>(xz, conv_w, conv_b, xs_f32, xs_bf);

    // 6) dbc = xs @ W_xproj   (M=16384, N=64, K=1024) -- WMMA, emits dt_r bf16
    gemm_bf16_kernel<<<dim3(NTOK / 128, DBCN / 64), 256, 0, stream>>>(
        xs_bf, Wxp_bf, dbc, dtr_bf, nullptr, NTOK, DBCN, DINNER, 2);

    // 7) delta = softplus(dt_r @ W_dt + b_dt)  (M=16384, N=1024, K=32) -- WMMA
    gemm_bf16_kernel<<<dim3(NTOK / 128, DINNER / 64), 256, 0, stream>>>(
        dtr_bf, Wdt_bf, delta, nullptr, b_dt, NTOK, DINNER, DTRANK, 1);

    // 8) chunk-parallel selective scan (+ D term, SiLU(z) gate, masked pooling)
    scan_chunk_kernel<<<NSEQ * NCHUNK * (DINNER / 256), 256, 0, stream>>>(
        xz, xs_f32, dbc, delta, A_log, D_param, counts, Earr, Harr, Garr, S0arr);
    scan_combine_kernel<<<(NSEQ * DINNER + 255) / 256, 256, 0, stream>>>(
        Earr, Harr, Garr, S0arr, lengths, ypool);

    // 9) pooled = ypool @ W_outproj ; out = pooled @ W_out + b_out
    small_gemm_kernel<<<(NSEQ * DMODEL + 255) / 256, 256, 0, stream>>>(
        ypool, W_outproj, nullptr, pooled, DMODEL, DINNER);
    small_gemm_kernel<<<(NSEQ * DMODEL + 255) / 256, 256, 0, stream>>>(
        pooled, W_out, b_out, out, DMODEL, DMODEL);
}
